// SNN_15281493639795
// MI455X (gfx1250) — matrix operand order repaired and checked by reading, not compiled, not run on hardware
//
#include <hip/hip_runtime.h>

typedef _Float16 half_t;
typedef __attribute__((ext_vector_type(16))) _Float16 v16h;
typedef __attribute__((ext_vector_type(8)))  _Float16 v8h;
typedef __attribute__((ext_vector_type(8)))  float    v8f;

#define SEQ   100
#define BATCH 256
#define INF   784
#define INFP  800          // 784 padded to a multiple of 32
#define H1    1024
#define H2    1024
#define NOUT  10
#define NOUTP 16

#define VTH   0.1f
#define SYNF  0.8f         // 1 - dt * tau_syn_inv
#define MEMF  0.1f         // dt * tau_mem_inv

#define GRID  32
#define BLOCK 256
#define WPB   8            // wave32 waves per block -> 256 waves total

// ---------------------------------------------------------------- WMMA core
static __device__ __forceinline__ v8f wmma16(v16h a, v16h b, v8f c) {
  // D = A(16x32 f16) * B(32x16 f16) + C(16x16 f32)
  return __builtin_amdgcn_wmma_f32_16x16x32_f16(false, a, false, b, (short)0, c,
                                                false, false);
}

// A fragment (16x32) from row-major f16 [rows, ld]; rows row0..row0+15, k0..k0+31.
// Lane (l<16): M=l, halves 0..15 = K k0..k0+15 (contiguous).
// Lane (l>=16): M=l-16, K k0+16..k0+31.  (matches B fragment k order)
static __device__ __forceinline__ v16h load_a_f16(const half_t* __restrict__ A,
                                                  int row0, int ld, int k0, int lane) {
  int m   = lane & 15;
  int off = (lane >> 4) << 4;
  const half_t* p = A + (size_t)(row0 + m) * ld + k0 + off;
  v8h lo = *(const v8h*)p;
  v8h hi = *(const v8h*)(p + 8);
  v16h r;
#pragma unroll
  for (int i = 0; i < 8; ++i) { r[i] = lo[i]; r[8 + i] = hi[i]; }
  return r;
}

// B fragment (32x16): B[k][n] = W[col0+n][k0+k] with W row-major [N, K].
// Lane (l<16): N=l, halves = K k0..k0+15 (one contiguous 32B run);
// lane (l>=16): N=l-16, K k0+16..k0+31.  (contiguous k order, same as A fragment)
static __device__ __forceinline__ v16h load_b_f16(const half_t* __restrict__ W,
                                                  int col0, int ld, int k0, int lane) {
  int n  = lane & 15;
  int kk = k0 + ((lane >> 4) << 4);
  const half_t* p = W + (size_t)(col0 + n) * ld + kk;
  __builtin_prefetch(p + 64, 0, 1);   // pull upcoming chunks of this weight row
  v8h lo = *(const v8h*)p;
  v8h hi = *(const v8h*)(p + 8);
  v16h r;
#pragma unroll
  for (int i = 0; i < 8; ++i) { r[i] = lo[i]; r[8 + i] = hi[i]; }
  return r;
}

// C/D layout: VGPR v, lane l  ->  M = v + 8*(l/16), N = l%16.
static __device__ __forceinline__ void lif_update(v8f acc, float* V, float* I,
                                                  half_t* Zn, int row0, int col0,
                                                  int ld, int lane) {
  int n  = lane & 15;
  int mh = (lane >> 4) << 3;
#pragma unroll
  for (int v = 0; v < 8; ++v) {
    size_t idx = (size_t)(row0 + mh + v) * ld + (col0 + n);
    float vv = V[idx], ii = I[idx];
    float vd = vv + MEMF * (ii - vv);     // v + mem*(-v + i)
    float id = ii * SYNF;
    float z  = (vd - VTH > 0.f) ? 1.f : 0.f;
    V[idx] = (1.f - z) * vd;
    I[idx] = id + acc[v];
    Zn[idx] = (half_t)z;
  }
}

static __device__ __forceinline__ void store_h16(v8f acc, half_t* __restrict__ H,
                                                 int row0, int col0, int lane) {
  int n  = lane & 15;
  int mh = (lane >> 4) << 3;
#pragma unroll
  for (int v = 0; v < 8; ++v)
    H[(size_t)(row0 + mh + v) * H2 + col0 + n] = (half_t)acc[v];
}

// Device-wide barrier: monotone counter (zeroed by memset before launch).
static __device__ __forceinline__ void grid_barrier(unsigned* bar, unsigned nblk) {
  __threadfence();            // release our writes (agent scope)
  __syncthreads();
  if (threadIdx.x == 0) {
    unsigned my = __hip_atomic_fetch_add(bar, 1u, __ATOMIC_ACQ_REL,
                                         __HIP_MEMORY_SCOPE_AGENT);
    unsigned target = (my / nblk + 1u) * nblk;
    while (__hip_atomic_load(bar, __ATOMIC_ACQUIRE, __HIP_MEMORY_SCOPE_AGENT) < target)
      __builtin_amdgcn_s_sleep(2);
  }
  __syncthreads();
  __threadfence();            // acquire: drop stale near-cache lines
}

// ------------------------------------------------------- setup: f32 -> f16 pad
__global__ void convert_pad_f16(const float* __restrict__ src, half_t* __restrict__ dst,
                                int src_rows, int src_cols, int dst_rows, int dst_cols) {
  int total = dst_rows * dst_cols;
  for (int idx = blockIdx.x * blockDim.x + threadIdx.x; idx < total;
       idx += gridDim.x * blockDim.x) {
    int r = idx / dst_cols, c = idx - r * dst_cols;
    float v = (r < src_rows && c < src_cols) ? src[(size_t)r * src_cols + c] : 0.f;
    dst[idx] = (half_t)v;
  }
}

// ------------------------------------------------------- persistent SNN scan
// 256 waves; each owns a 32x32 output tile (2x2 WMMA register tiling).
__global__ __launch_bounds__(BLOCK)
void snn_persistent(const half_t* __restrict__ x16,
                    const half_t* __restrict__ w1_in, const half_t* __restrict__ w1_rec,
                    const half_t* __restrict__ wh,   const half_t* __restrict__ w2_in,
                    const half_t* __restrict__ w2_rec, const half_t* __restrict__ wo,
                    half_t* z1a, half_t* z1b, half_t* z2a, half_t* z2b, half_t* hbuf,
                    float* v1, float* i1, float* v2, float* i2, float* vo, float* io,
                    float* __restrict__ out, unsigned* bar) {
  const int lane = threadIdx.x & 31;
  const int wid  = blockIdx.x * WPB + (threadIdx.x >> 5);   // 0..255
  const int row0 = (wid >> 5) << 5;                         // 8 batch super-tiles
  const int col0 = (wid & 31) << 5;                         // 32 neuron super-tiles
  const unsigned nblk = gridDim.x;

#define GEMM2x2(Asrc, lda, Wsrc, ldw, k0)                                   \
  {                                                                         \
    v16h a0 = load_a_f16(Asrc, row0,      lda, k0, lane);                   \
    v16h a1 = load_a_f16(Asrc, row0 + 16, lda, k0, lane);                   \
    v16h b0 = load_b_f16(Wsrc, col0,      ldw, k0, lane);                   \
    v16h b1 = load_b_f16(Wsrc, col0 + 16, ldw, k0, lane);                   \
    acc00 = wmma16(a0, b0, acc00);                                          \
    acc01 = wmma16(a0, b1, acc01);                                          \
    acc10 = wmma16(a1, b0, acc10);                                          \
    acc11 = wmma16(a1, b1, acc11);                                          \
  }

  for (int t = 0; t < SEQ; ++t) {
    const half_t* z1o = (t & 1) ? z1b : z1a;
    half_t*       z1n = (t & 1) ? z1a : z1b;
    const half_t* z2o = (t & 1) ? z2b : z2a;
    half_t*       z2n = (t & 1) ? z2a : z2b;
    const half_t* xt  = x16 + (size_t)t * BATCH * INFP;

    // Phase A: i1 = i1*syn + x@w1_in^T + z1_old@w1_rec^T ; LIF-1 -> z1_new
    {
      v8f acc00 = {0,0,0,0,0,0,0,0}, acc01 = {0,0,0,0,0,0,0,0};
      v8f acc10 = {0,0,0,0,0,0,0,0}, acc11 = {0,0,0,0,0,0,0,0};
      for (int k0 = 0; k0 < INFP; k0 += 32) GEMM2x2(xt,  INFP, w1_in,  INFP, k0);
      for (int k0 = 0; k0 < H1;   k0 += 32) GEMM2x2(z1o, H1,   w1_rec, H1,   k0);
      lif_update(acc00, v1, i1, z1n, row0,      col0,      H1, lane);
      lif_update(acc01, v1, i1, z1n, row0,      col0 + 16, H1, lane);
      lif_update(acc10, v1, i1, z1n, row0 + 16, col0,      H1, lane);
      lif_update(acc11, v1, i1, z1n, row0 + 16, col0 + 16, H1, lane);
    }
    grid_barrier(bar, nblk);

    // Phase B: h = z1_new @ wh^T
    {
      v8f acc00 = {0,0,0,0,0,0,0,0}, acc01 = {0,0,0,0,0,0,0,0};
      v8f acc10 = {0,0,0,0,0,0,0,0}, acc11 = {0,0,0,0,0,0,0,0};
      for (int k0 = 0; k0 < H1; k0 += 32) GEMM2x2(z1n, H1, wh, H1, k0);
      store_h16(acc00, hbuf, row0,      col0,      lane);
      store_h16(acc01, hbuf, row0,      col0 + 16, lane);
      store_h16(acc10, hbuf, row0 + 16, col0,      lane);
      store_h16(acc11, hbuf, row0 + 16, col0 + 16, lane);
    }
    grid_barrier(bar, nblk);

    // Phase C: i2 = i2*syn + h@w2_in^T + z2_old@w2_rec^T ; LIF-2 -> z2_new
    {
      v8f acc00 = {0,0,0,0,0,0,0,0}, acc01 = {0,0,0,0,0,0,0,0};
      v8f acc10 = {0,0,0,0,0,0,0,0}, acc11 = {0,0,0,0,0,0,0,0};
      for (int k0 = 0; k0 < H2; k0 += 32) GEMM2x2(hbuf, H2, w2_in,  H2, k0);
      for (int k0 = 0; k0 < H2; k0 += 32) GEMM2x2(z2o,  H2, w2_rec, H2, k0);
      lif_update(acc00, v2, i2, z2n, row0,      col0,      H2, lane);
      lif_update(acc01, v2, i2, z2n, row0,      col0 + 16, H2, lane);
      lif_update(acc10, v2, i2, z2n, row0 + 16, col0,      H2, lane);
      lif_update(acc11, v2, i2, z2n, row0 + 16, col0 + 16, H2, lane);
    }
    grid_barrier(bar, nblk);

    // Phase D (16 waves): o = z2_new @ wo^T ; LI cell ; record voltages.
    // Disjoint from next step's Phase A, so no trailing barrier needed.
    if (wid < 16) {
      int r0 = wid << 4;
      v8f acc = {0,0,0,0,0,0,0,0};
      for (int k0 = 0; k0 < H2; k0 += 32)
        acc = wmma16(load_a_f16(z2n, r0, H2, k0, lane),
                     load_b_f16(wo, 0, H2, k0, lane), acc);
      int n = lane & 15, mh = (lane >> 4) << 3;
#pragma unroll
      for (int v = 0; v < 8; ++v) {
        int bb = r0 + mh + v;
        float vo_ = vo[bb * NOUTP + n];
        float io_ = io[bb * NOUTP + n];
        float vn  = vo_ + MEMF * (io_ - vo_);
        vo[bb * NOUTP + n] = vn;
        io[bb * NOUTP + n] = io_ * SYNF + acc[v];
        if (n < NOUT)
          out[((size_t)t * BATCH + bb) * NOUT + n] = vn;
      }
    }
  }
#undef GEMM2x2
}

// ----------------------------------------------------------------- launcher
extern "C" void kernel_launch(void* const* d_in, const int* in_sizes, int n_in,
                              void* d_out, int out_size, void* d_ws, size_t ws_size,
                              hipStream_t stream) {
  (void)in_sizes; (void)n_in; (void)out_size; (void)ws_size;
  const float* x      = (const float*)d_in[0];
  const float* w1_in  = (const float*)d_in[1];
  const float* w1_rec = (const float*)d_in[2];
  const float* wh     = (const float*)d_in[3];
  const float* w2_in  = (const float*)d_in[4];
  const float* w2_rec = (const float*)d_in[5];
  const float* wo     = (const float*)d_in[6];
  float* out = (float*)d_out;

  char* ws = (char*)d_ws;
  size_t off = 0;
  auto carve = [&](size_t bytes) -> char* {
    char* p = ws + off;
    off += (bytes + 255) & ~(size_t)255;
    return p;
  };
  unsigned* bar      = (unsigned*)carve(256);
  half_t*   x16      = (half_t*)carve((size_t)SEQ * BATCH * INFP * 2);
  half_t*   w1_in16  = (half_t*)carve((size_t)H1 * INFP * 2);
  half_t*   w1_rec16 = (half_t*)carve((size_t)H1 * H1 * 2);
  half_t*   wh16     = (half_t*)carve((size_t)H2 * H1 * 2);
  half_t*   w2_in16  = (half_t*)carve((size_t)H2 * H2 * 2);
  half_t*   w2_rec16 = (half_t*)carve((size_t)H2 * H2 * 2);
  half_t*   wo16     = (half_t*)carve((size_t)NOUTP * H2 * 2);
  char* state_base = ws + off;
  half_t* z1a  = (half_t*)carve((size_t)BATCH * H1 * 2);
  half_t* z1b  = (half_t*)carve((size_t)BATCH * H1 * 2);
  half_t* z2a  = (half_t*)carve((size_t)BATCH * H2 * 2);
  half_t* z2b  = (half_t*)carve((size_t)BATCH * H2 * 2);
  half_t* hbuf = (half_t*)carve((size_t)BATCH * H2 * 2);
  float*  v1   = (float*)carve((size_t)BATCH * H1 * 4);
  float*  i1   = (float*)carve((size_t)BATCH * H1 * 4);
  float*  v2   = (float*)carve((size_t)BATCH * H2 * 4);
  float*  i2   = (float*)carve((size_t)BATCH * H2 * 4);
  float*  vo   = (float*)carve((size_t)BATCH * NOUTP * 4);
  float*  io   = (float*)carve((size_t)BATCH * NOUTP * 4);
  size_t state_bytes = (size_t)((ws + off) - state_base);

  // Reset barrier + all carried state every call (graph-replay determinism).
  hipMemsetAsync(bar, 0, 256, stream);
  hipMemsetAsync(state_base, 0, state_bytes, stream);

  auto cgrid = [](size_t total) {
    size_t g = (total + 255) / 256;
    return (int)(g > 4096 ? 4096 : g);
  };
  // Inputs/weights -> padded f16 (spikes {0,1} are exact in f16).
  convert_pad_f16<<<cgrid((size_t)SEQ * BATCH * INFP), 256, 0, stream>>>(
      x, x16, SEQ * BATCH, INF, SEQ * BATCH, INFP);
  convert_pad_f16<<<cgrid((size_t)H1 * INFP), 256, 0, stream>>>(w1_in,  w1_in16,  H1,   INF, H1,    INFP);
  convert_pad_f16<<<cgrid((size_t)H1 * H1),   256, 0, stream>>>(w1_rec, w1_rec16, H1,   H1,  H1,    H1);
  convert_pad_f16<<<cgrid((size_t)H2 * H1),   256, 0, stream>>>(wh,     wh16,     H2,   H1,  H2,    H1);
  convert_pad_f16<<<cgrid((size_t)H2 * H2),   256, 0, stream>>>(w2_in,  w2_in16,  H2,   H2,  H2,    H2);
  convert_pad_f16<<<cgrid((size_t)H2 * H2),   256, 0, stream>>>(w2_rec, w2_rec16, H2,   H2,  H2,    H2);
  convert_pad_f16<<<cgrid((size_t)NOUTP * H2),256, 0, stream>>>(wo,     wo16,     NOUT, H2,  NOUTP, H2);

  snn_persistent<<<GRID, BLOCK, 0, stream>>>(x16, w1_in16, w1_rec16, wh16, w2_in16,
                                             w2_rec16, wo16, z1a, z1b, z2a, z2b,
                                             hbuf, v1, i1, v2, i2, vo, io, out, bar);
}
